// AlternatingDecoderFasttext_53755810677293
// MI455X (gfx1250) — matrix operand order repaired and checked by reading, not compiled, not run on hardware
//
#include <hip/hip_runtime.h>

// ---------------------------------------------------------------------------
// AlternatingDecoderFasttext for MI455X (gfx1250, wave32, WMMA bf16)
//   B=16 S=128 D=768 H=12 dk=64 DFF=3072 NL=6 SRC=256
// - All matmuls on V_WMMA_F32_16X16X32_BF16 (fp32 accumulate).
// - Weights pre-transposed+converted to bf16 [N][K] once per launch, then
//   streamed into LDS with GLOBAL_LOAD_ASYNC_TO_LDS_B128 (ASYNCcnt path).
// - Attention is a flash-style streaming kernel (online softmax).
// ---------------------------------------------------------------------------

#define B_   16
#define S_   128
#define D_   768
#define H_   12
#define DK_  64
#define DFF_ 3072
#define NL_  6
#define SRC_ 256
#define NTOK_ (B_ * S_)   // 2048

typedef __attribute__((ext_vector_type(16))) __bf16          v16bf;
typedef __attribute__((ext_vector_type(16))) unsigned short  v16u;
typedef __attribute__((ext_vector_type(8)))  float           v8f;

__device__ __forceinline__ unsigned short f2bf(float f) {
  unsigned int u = __float_as_uint(f);
  unsigned int r = u + 0x7FFFu + ((u >> 16) & 1u);   // round-to-nearest-even
  return (unsigned short)(r >> 16);
}

// A-fragment (16x32 bf16): lane&15 = M row; lane bit4 selects the
// K-half (+16); reg r pairs: K = kh + r*2 (matches lds_bfrag).
__device__ __forceinline__ v16bf lds_afrag(const unsigned short* s, int stride,
                                           int m0, int k0) {
  const int lane = threadIdx.x & 31;
  const int m  = m0 + (lane & 15);
  const int kh = (lane & 16) ? 16 : 0;
  const unsigned short* row = s + m * stride + k0;
  v16u u;
#pragma unroll
  for (int r = 0; r < 8; ++r) {
    const int kb = kh + r * 2;
    u[2 * r]     = row[kb];
    u[2 * r + 1] = row[kb + 1];
  }
  return __builtin_bit_cast(v16bf, u);
}

// B-fragment (32x16 bf16): lane&15 = N col; lanes 0-15 hold K=0..15, lanes
// 16-31 hold K=16..31. LDS stored [N][K] so 16 K-values are contiguous.
__device__ __forceinline__ v16bf lds_bfrag(const unsigned short* s, int stride,
                                           int n0, int k0) {
  const int lane = threadIdx.x & 31;
  const int n  = n0 + (lane & 15);
  const int kh = (lane & 16) ? 16 : 0;
  const v16u* p = (const v16u*)(s + n * stride + k0 + kh);
  return __builtin_bit_cast(v16bf, *p);
}

__device__ __forceinline__ v8f wmma_bf16(v16bf a, v16bf b, v8f c) {
  return __builtin_amdgcn_wmma_f32_16x16x32_bf16(false, a, false, b,
                                                 (short)0, c, false, false);
}

// LDS byte offset of a __shared__ object (flat LDS aperture keeps the
// workgroup-relative offset in addr[31:0]).
__device__ __forceinline__ unsigned lds_off(const void* p) {
  return (unsigned)(unsigned long long)(uintptr_t)p;
}

// ---------------------------------------------------------------------------
// Embedding + positional encoding; also broadcasts y_param into y.
// ---------------------------------------------------------------------------
__global__ __launch_bounds__(256) void embed_kernel(
    const int* __restrict__ trg, const float* __restrict__ emb,
    const float* __restrict__ yparam, const float* __restrict__ pe_alpha,
    float* __restrict__ X, float* __restrict__ Y) {
  const int row = blockIdx.x;           // b*S + s
  const int s   = row & (S_ - 1);
  const int tok = trg[row];
  const float al = pe_alpha[0];
  const float c  = -logf(10000.0f) / (float)D_;
#pragma unroll
  for (int j = 0; j < 3; ++j) {
    const int d = threadIdx.x + j * 256;
    const int half = d >> 1;
    const float div = expf(c * (float)(2 * half));
    const float ang = (float)s * div;
    const float pe  = (d & 1) ? cosf(ang) : sinf(ang);
    X[(size_t)row * D_ + d] = emb[(size_t)tok * D_ + d] + al * pe;
    Y[(size_t)row * D_ + d] = yparam[(size_t)s * D_ + d];
  }
}

// ---------------------------------------------------------------------------
// Combined-mask cutoff r[b] (cummin with sentinel, indexed at min(b, B-2)).
// ---------------------------------------------------------------------------
__global__ void rowcut_kernel(const int* __restrict__ trg, int* __restrict__ r) {
  __shared__ int c[B_];
  const int b = threadIdx.x;
  if (b < B_) {
    int start = 0;
    for (int s = 0; s < S_; ++s) {
      if (trg[b * S_ + s] == 0) { start = s; break; }
    }
    c[b] = (start == 0) ? (S_ + 1) : start;
  }
  __syncthreads();
  if (threadIdx.x == 0) {
    int cmin[B_];
    int cm = 0x7fffffff;
    for (int j = 0; j < B_; ++j) { cm = (c[j] < cm) ? c[j] : cm; cmin[j] = cm; }
    for (int bb = 0; bb < B_; ++bb) {
      const int idx = (bb < B_ - 2) ? bb : (B_ - 2);
      r[bb] = cmin[idx];
    }
  }
}

// ---------------------------------------------------------------------------
// LayerNorm, one 768-wide row per block.
// ---------------------------------------------------------------------------
__global__ __launch_bounds__(256) void ln_kernel(
    const float* __restrict__ X, const float* __restrict__ ga,
    const float* __restrict__ be, float* __restrict__ Y) {
  __shared__ float red[256];
  const int tid = threadIdx.x;
  const float* x = X + (size_t)blockIdx.x * D_;
  float v0 = x[tid], v1 = x[tid + 256], v2 = x[tid + 512];
  red[tid] = v0 + v1 + v2;
  __syncthreads();
  for (int o = 128; o > 0; o >>= 1) {
    if (tid < o) red[tid] += red[tid + o];
    __syncthreads();
  }
  const float mean = red[0] * (1.0f / D_);
  __syncthreads();
  const float d0 = v0 - mean, d1 = v1 - mean, d2 = v2 - mean;
  red[tid] = d0 * d0 + d1 * d1 + d2 * d2;
  __syncthreads();
  for (int o = 128; o > 0; o >>= 1) {
    if (tid < o) red[tid] += red[tid + o];
    __syncthreads();
  }
  const float inv = rsqrtf(red[0] * (1.0f / D_) + 1e-6f);
  float* y = Y + (size_t)blockIdx.x * D_;
  y[tid]       = ga[tid]       * d0 * inv + be[tid];
  y[tid + 256] = ga[tid + 256] * d1 * inv + be[tid + 256];
  y[tid + 512] = ga[tid + 512] * d2 * inv + be[tid + 512];
}

// ---------------------------------------------------------------------------
// Weight transpose + fp32->bf16: W[K][N] -> Wt[N][K], tiled through LDS so
// both global phases are coalesced. grid = (N/32, K/32), block 256 (32x8).
// ---------------------------------------------------------------------------
__global__ __launch_bounds__(256) void wt_kernel(
    const float* __restrict__ W, unsigned short* __restrict__ Wt,
    int K, int N) {
  __shared__ unsigned short t[32][33];
  const int tx = threadIdx.x & 31, ty = threadIdx.x >> 5;
  const int kt = blockIdx.y * 32, nt = blockIdx.x * 32;
#pragma unroll
  for (int r = 0; r < 4; ++r)
    t[ty + r * 8][tx] = f2bf(W[(size_t)(kt + ty + r * 8) * N + nt + tx]);
  __syncthreads();
#pragma unroll
  for (int r = 0; r < 4; ++r)
    Wt[(size_t)(nt + ty + r * 8) * K + kt + tx] = t[tx][ty + r * 8];
}

// ---------------------------------------------------------------------------
// GEMM: C[M,N] = (resid?) + act( A[M,K] @ W[K,N] + bias[N] )
// Weights pre-converted bf16, layout Wt[N][K]. Block tile 64x256, 8 waves of
// 32x64 (2x4 WMMA frags, 8 wmma per K-step). Weight tile streamed to LDS via
// GLOBAL_LOAD_ASYNC_TO_LDS_B128 (64 B per thread per step).
// M%64==0, N%256==0, K%32==0 at every call site.
// ---------------------------------------------------------------------------
__global__ __launch_bounds__(256) void gemm_kernel(
    const float* __restrict__ A, const unsigned short* __restrict__ Wt,
    const float* __restrict__ bias, const float* resid, float* C,
    int M, int N, int K, int relu) {
  __shared__ __align__(32) unsigned short sA[64 * 32];    // [m][k]  4 KB
  __shared__ __align__(32) unsigned short sB[256 * 32];   // [n][k] 16 KB
  const int tid  = threadIdx.x;
  const int lane = tid & 31;
  const int wave = tid >> 5;
  const int m0 = blockIdx.y * 64;
  const int n0 = blockIdx.x * 256;
  const int wm = (wave >> 2) * 32;   // 0 / 32
  const int wn = (wave & 3) * 64;    // 0..192

  v8f acc[2][4];
#pragma unroll
  for (int ni = 0; ni < 4; ++ni) {
    const float bv = bias ? bias[n0 + wn + ni * 16 + (lane & 15)] : 0.0f;
#pragma unroll
    for (int mi = 0; mi < 2; ++mi)
#pragma unroll
      for (int r = 0; r < 8; ++r) acc[mi][ni][r] = bv;
  }

  const unsigned           sBoff = lds_off(sB);
  const unsigned long long wbase = (unsigned long long)(uintptr_t)(const void*)Wt;

  for (int kk = 0; kk < K; kk += 32) {
    // --- async weight tile: thread t owns row n0+t (32 bf16 = 64 B) ---
    {
      const unsigned long long ga =
          wbase + 2ull * ((unsigned long long)(n0 + tid) * (unsigned)K + (unsigned)kk);
      const unsigned la = sBoff + (unsigned)tid * 64u;
#pragma unroll
      for (int seg = 0; seg < 4; ++seg) {
        asm volatile("global_load_async_to_lds_b128 %0, %1, off"
                     :: "v"(la + (unsigned)(seg * 16)),
                        "v"(ga + (unsigned long long)(seg * 16))
                     : "memory");
      }
    }
    // --- activation tile fp32 -> bf16 through VGPRs ---
    for (int i = tid; i < 64 * 32; i += 256) {
      const int m = i >> 5, k = i & 31;
      sA[i] = f2bf(A[(size_t)(m0 + m) * K + kk + k]);
    }
    asm volatile("s_wait_asynccnt 0" ::: "memory");
    __syncthreads();

    const v16bf a0 = lds_afrag(sA, 32, wm, 0);
    const v16bf a1 = lds_afrag(sA, 32, wm + 16, 0);
#pragma unroll
    for (int ni = 0; ni < 4; ++ni) {
      const v16bf bfr = lds_bfrag(sB, 32, wn + ni * 16, 0);
      acc[0][ni] = wmma_bf16(a0, bfr, acc[0][ni]);
      acc[1][ni] = wmma_bf16(a1, bfr, acc[1][ni]);
    }
    __syncthreads();
  }

#pragma unroll
  for (int mi = 0; mi < 2; ++mi)
#pragma unroll
    for (int ni = 0; ni < 4; ++ni) {
      const int mbase = m0 + wm + mi * 16 + ((lane & 16) ? 8 : 0);
      const int n     = n0 + wn + ni * 16 + (lane & 15);
#pragma unroll
      for (int r = 0; r < 8; ++r) {
        float v = acc[mi][ni][r];
        if (relu)  v = fmaxf(v, 0.0f);
        if (resid) v += resid[(size_t)(mbase + r) * N + n];
        C[(size_t)(mbase + r) * N + n] = v;
      }
    }
}

// ---------------------------------------------------------------------------
// Flash attention: one block per (b,h). Sq=128, dk=64, Sk in {128,256}.
// mode: 0 = no mask, 1 = causal + rowcut[b], 2 = src_mask[b,k].
// ---------------------------------------------------------------------------
__global__ __launch_bounds__(256) void flash_kernel(
    const float* __restrict__ Q, const float* __restrict__ Kp,
    const float* __restrict__ Vp, float* __restrict__ O,
    const int* __restrict__ rowcut, const int* __restrict__ srcmask,
    int Sk, int mode) {
  __shared__ __align__(32) unsigned short sQ[S_ * DK_];    // [q][d]   16 KB
  __shared__ __align__(32) unsigned short sKP[S_ * DK_];   // Kc / Pc  16 KB
  __shared__ __align__(32) unsigned short sVt[DK_ * 64];   // [d][key]  8 KB
  const int tid  = threadIdx.x;
  const int lane = tid & 31;
  const int wave = tid >> 5;
  const int b = blockIdx.x / H_;
  const int h = blockIdx.x % H_;
  const float scale = 0.125f;   // 1/sqrt(64)

  for (int i = tid; i < S_ * DK_; i += 256) {
    const int s = i >> 6, d = i & 63;
    sQ[i] = f2bf(Q[(size_t)(b * S_ + s) * D_ + h * DK_ + d]);
  }

  v8f o[4];
#pragma unroll
  for (int t = 0; t < 4; ++t)
#pragma unroll
    for (int r = 0; r < 8; ++r) o[t][r] = 0.0f;
  float mrow[8], lrow[8];
#pragma unroll
  for (int r = 0; r < 8; ++r) { mrow[r] = -3.0e38f; lrow[r] = 0.0f; }
  const int rb = (mode == 1) ? rowcut[b] : 0;
  const int qb = wave * 16 + ((lane & 16) ? 8 : 0);

  for (int c = 0; c < Sk; c += 64) {
    __syncthreads();   // prior chunk's Pc / Vt reads done before overwrite
    for (int i = tid; i < 64 * 64; i += 256) {
      const int key = i >> 6, d = i & 63;
      sKP[key * 64 + d] = f2bf(Kp[(size_t)(b * Sk + c + key) * D_ + h * DK_ + d]);
    }
    for (int i = tid; i < 64 * 64; i += 256) {
      const int key = i >> 6, d = i & 63;                  // coalesced over d
      sVt[d * 64 + key] = f2bf(Vp[(size_t)(b * Sk + c + key) * D_ + h * DK_ + d]);
    }
    __syncthreads();

    const v16bf a0 = lds_afrag(sQ, 64, wave * 16, 0);
    const v16bf a1 = lds_afrag(sQ, 64, wave * 16, 32);
    v8f sf[4];
#pragma unroll
    for (int nt = 0; nt < 4; ++nt) {
      v8f z;
#pragma unroll
      for (int r = 0; r < 8; ++r) z[r] = 0.0f;
      const v16bf kb0 = lds_bfrag(sKP, 64, nt * 16, 0);
      const v16bf kb1 = lds_bfrag(sKP, 64, nt * 16, 32);
      z = wmma_bf16(a0, kb0, z);
      z = wmma_bf16(a1, kb1, z);
      sf[nt] = z;
    }

#pragma unroll
    for (int nt = 0; nt < 4; ++nt) {
      const int kidx = c + nt * 16 + (lane & 15);
      int keepsrc = 1;
      if (mode == 2) keepsrc = srcmask[b * SRC_ + kidx];
#pragma unroll
      for (int r = 0; r < 8; ++r) {
        float v = sf[nt][r] * scale;
        if (mode == 1) {
          const int q = qb + r;
          if (!(kidx <= q && q < rb)) v = -1e9f;
        } else if (mode == 2) {
          if (!keepsrc) v = -1e9f;
        }
        sf[nt][r] = v;
      }
    }

    // online softmax (a row lives across the 16 lanes of a half-wave)
    float nm[8], alpha[8];
#pragma unroll
    for (int r = 0; r < 8; ++r) {
      float cm = fmaxf(fmaxf(sf[0][r], sf[1][r]), fmaxf(sf[2][r], sf[3][r]));
      for (int off = 1; off < 16; off <<= 1)
        cm = fmaxf(cm, __shfl_xor(cm, off, 32));
      nm[r]    = fmaxf(mrow[r], cm);
      alpha[r] = __expf(mrow[r] - nm[r]);
    }
#pragma unroll
    for (int nt = 0; nt < 4; ++nt)
#pragma unroll
      for (int r = 0; r < 8; ++r) sf[nt][r] = __expf(sf[nt][r] - nm[r]);
#pragma unroll
    for (int r = 0; r < 8; ++r) {
      float cs = sf[0][r] + sf[1][r] + sf[2][r] + sf[3][r];
      for (int off = 1; off < 16; off <<= 1) cs += __shfl_xor(cs, off, 32);
      lrow[r] = lrow[r] * alpha[r] + cs;
      mrow[r] = nm[r];
    }
#pragma unroll
    for (int t = 0; t < 4; ++t)
#pragma unroll
      for (int r = 0; r < 8; ++r) o[t][r] *= alpha[r];

    __syncthreads();   // all waves done reading Kc; Pc overwrites it
#pragma unroll
    for (int nt = 0; nt < 4; ++nt) {
      const int k = nt * 16 + (lane & 15);
#pragma unroll
      for (int r = 0; r < 8; ++r) sKP[(qb + r) * 64 + k] = f2bf(sf[nt][r]);
    }
    __syncthreads();   // Pc visible

    const v16bf p0 = lds_afrag(sKP, 64, wave * 16, 0);
    const v16bf p1 = lds_afrag(sKP, 64, wave * 16, 32);
#pragma unroll
    for (int dt = 0; dt < 4; ++dt) {
      const v16bf vb0 = lds_bfrag(sVt, 64, dt * 16, 0);
      const v16bf vb1 = lds_bfrag(sVt, 64, dt * 16, 32);
      o[dt] = wmma_bf16(p0, vb0, o[dt]);
      o[dt] = wmma_bf16(p1, vb1, o[dt]);
    }
  }

#pragma unroll
  for (int dt = 0; dt < 4; ++dt) {
    const int d = dt * 16 + (lane & 15);
#pragma unroll
    for (int r = 0; r < 8; ++r) {
      const int q = qb + r;
      O[(size_t)(b * S_ + q) * D_ + h * DK_ + d] = o[dt][r] / lrow[r];
    }
  }
}

// ---------------------------------------------------------------------------
// Host orchestration.
// ---------------------------------------------------------------------------
extern "C" void kernel_launch(void* const* d_in, const int* in_sizes, int n_in,
                              void* d_out, int out_size, void* d_ws,
                              size_t ws_size, hipStream_t stream) {
  (void)in_sizes; (void)n_in; (void)out_size; (void)ws_size;

  const int*   trg      = (const int*)  d_in[0];
  const float* e_out    = (const float*)d_in[1];
  const int*   src_mask = (const int*)  d_in[2];
  const float* emb      = (const float*)d_in[3];
  const float* y_param  = (const float*)d_in[4];
  const float* pe_alpha = (const float*)d_in[5];
  const float* Wq       = (const float*)d_in[6];
  const float* Wk       = (const float*)d_in[7];
  const float* Wv       = (const float*)d_in[8];
  const float* Wo       = (const float*)d_in[9];
  const float* bq       = (const float*)d_in[10];
  const float* bk       = (const float*)d_in[11];
  const float* bv       = (const float*)d_in[12];
  const float* bo       = (const float*)d_in[13];
  const float* W1       = (const float*)d_in[14];
  const float* b1       = (const float*)d_in[15];
  const float* W2       = (const float*)d_in[16];
  const float* b2       = (const float*)d_in[17];
  const float* ln_a     = (const float*)d_in[18];
  const float* ln_b     = (const float*)d_in[19];
  const float* final_a  = (const float*)d_in[20];
  const float* final_b  = (const float*)d_in[21];

  float* ws = (float*)d_ws;
  float* x   = ws;
  float* y   = x   + (size_t)NTOK_ * D_;
  float* x2  = y   + (size_t)NTOK_ * D_;
  float* y3  = x2  + (size_t)NTOK_ * D_;
  float* qb_ = y3  + (size_t)NTOK_ * D_;
  float* kb_ = qb_ + (size_t)NTOK_ * D_;
  float* vb_ = kb_ + (size_t)(B_ * SRC_) * D_;
  float* ab_ = vb_ + (size_t)(B_ * SRC_) * D_;
  float* hb_ = ab_ + (size_t)NTOK_ * D_;
  int*   rcut = (int*)(hb_ + (size_t)NTOK_ * DFF_);
  unsigned short* wt = (unsigned short*)(rcut + 64);

  const size_t WDD = (size_t)D_ * D_;             // 589824
  const size_t WDF = (size_t)D_ * DFF_;           // 2359296
  const size_t LAYER_W = 12 * WDD + 2 * WDF;      // bf16 elems per layer
  auto wtq = [&](int i, int j) { return wt + i * LAYER_W + (size_t)(0 + j) * WDD; };
  auto wtk = [&](int i, int j) { return wt + i * LAYER_W + (size_t)(3 + j) * WDD; };
  auto wtv = [&](int i, int j) { return wt + i * LAYER_W + (size_t)(6 + j) * WDD; };
  auto wto = [&](int i, int j) { return wt + i * LAYER_W + (size_t)(9 + j) * WDD; };
  auto wt1 = [&](int i) { return wt + i * LAYER_W + 12 * WDD; };
  auto wt2 = [&](int i) { return wt + i * LAYER_W + 12 * WDD + WDF; };

  auto tr = [&](const float* W, unsigned short* Wt, int K, int N) {
    dim3 g(N / 32, K / 32);
    wt_kernel<<<g, 256, 0, stream>>>(W, Wt, K, N);
  };
  auto gemm = [&](const float* A, const unsigned short* Wt, const float* bias,
                  const float* resid, float* C, int M, int N, int K, int relu) {
    dim3 grid(N / 256, M / 64);
    gemm_kernel<<<grid, 256, 0, stream>>>(A, Wt, bias, resid, C, M, N, K, relu);
  };
  auto ln = [&](const float* X, const float* ga, const float* be, float* Y) {
    ln_kernel<<<NTOK_, 256, 0, stream>>>(X, ga, be, Y);
  };

  // one-time weight transpose+convert (every call; deterministic)
  for (int i = 0; i < NL_; ++i) {
    for (int j = 0; j < 3; ++j) {
      tr(Wq + (size_t)(i * 3 + j) * WDD, wtq(i, j), D_, D_);
      tr(Wk + (size_t)(i * 3 + j) * WDD, wtk(i, j), D_, D_);
      tr(Wv + (size_t)(i * 3 + j) * WDD, wtv(i, j), D_, D_);
      tr(Wo + (size_t)(i * 3 + j) * WDD, wto(i, j), D_, D_);
    }
    tr(W1 + (size_t)i * WDF, wt1(i), D_, DFF_);
    tr(W2 + (size_t)i * WDF, wt2(i), DFF_, D_);
  }

  embed_kernel<<<NTOK_, 256, 0, stream>>>(trg, emb, y_param, pe_alpha, x, y);
  rowcut_kernel<<<1, 32, 0, stream>>>(trg, rcut);

  for (int i = 0; i < NL_; ++i) {
    const float* bq_i = bq + (size_t)(i * 3) * D_;
    const float* bk_i = bk + (size_t)(i * 3) * D_;
    const float* bv_i = bv + (size_t)(i * 3) * D_;
    const float* bo_i = bo + (size_t)(i * 3) * D_;
    const float* lna  = ln_a + (size_t)(i * 5) * D_;
    const float* lnb  = ln_b + (size_t)(i * 5) * D_;

    // --- self-attention (causal + rowcut mask) ---
    ln(x, lna + 0 * D_, lnb + 0 * D_, x2);
    gemm(x2, wtq(i, 0), bq_i + 0 * D_, nullptr, qb_, NTOK_, D_, D_, 0);
    gemm(x2, wtk(i, 0), bk_i + 0 * D_, nullptr, kb_, NTOK_, D_, D_, 0);
    gemm(x2, wtv(i, 0), bv_i + 0 * D_, nullptr, vb_, NTOK_, D_, D_, 0);
    flash_kernel<<<B_ * H_, 256, 0, stream>>>(qb_, kb_, vb_, ab_, rcut, nullptr,
                                              S_, 1);
    gemm(ab_, wto(i, 0), bo_i + 0 * D_, x, x, NTOK_, D_, D_, 0);

    // --- cross-attention vs encoder outputs (src mask) ---
    ln(y, lna + 1 * D_, lnb + 1 * D_, x2);   // y2 (reuse x2 buffer)
    gemm(x2, wtq(i, 1), bq_i + 1 * D_, nullptr, qb_, NTOK_, D_, D_, 0);
    gemm(e_out, wtk(i, 1), bk_i + 1 * D_, nullptr, kb_, B_ * SRC_, D_, D_, 0);
    gemm(e_out, wtv(i, 1), bv_i + 1 * D_, nullptr, vb_, B_ * SRC_, D_, D_, 0);
    flash_kernel<<<B_ * H_, 256, 0, stream>>>(qb_, kb_, vb_, ab_, nullptr,
                                              src_mask, SRC_, 2);
    gemm(ab_, wto(i, 1), bo_i + 1 * D_, y, y, NTOK_, D_, D_, 0);

    // --- x attends to y (no mask) ---
    ln(x, lna + 2 * D_, lnb + 2 * D_, x2);
    ln(y, lna + 3 * D_, lnb + 3 * D_, y3);
    gemm(x2, wtq(i, 2), bq_i + 2 * D_, nullptr, qb_, NTOK_, D_, D_, 0);
    gemm(y3, wtk(i, 2), bk_i + 2 * D_, nullptr, kb_, NTOK_, D_, D_, 0);
    gemm(y3, wtv(i, 2), bv_i + 2 * D_, nullptr, vb_, NTOK_, D_, D_, 0);
    flash_kernel<<<B_ * H_, 256, 0, stream>>>(qb_, kb_, vb_, ab_, nullptr,
                                              nullptr, S_, 0);
    gemm(ab_, wto(i, 2), bo_i + 2 * D_, x, x, NTOK_, D_, D_, 0);

    // --- FFN ---
    ln(x, lna + 4 * D_, lnb + 4 * D_, x2);
    gemm(x2, wt1(i), b1 + (size_t)i * DFF_, nullptr, hb_, NTOK_, DFF_, D_, 1);
    gemm(hb_, wt2(i), b2 + (size_t)i * D_, x, x, NTOK_, D_, DFF_, 0);
  }

  ln_kernel<<<NTOK_, 256, 0, stream>>>(x, final_a, final_b, (float*)d_out);
}
